// SwinTransformerBlock3D_2473901163193
// MI455X (gfx1250) — matrix operand order repaired and checked by reading, not compiled, not run on hardware
//
#include <hip/hip_runtime.h>
#include <hip/hip_bf16.h>
#include <math.h>

typedef _Float16 half_t;
typedef __attribute__((ext_vector_type(16))) _Float16 v16h;
typedef __attribute__((ext_vector_type(8)))  float    v8f;

#define WMMA_F16(a,b,c) __builtin_amdgcn_wmma_f32_16x16x32_f16(false,(a),false,(b),(short)0,(c),false,false)

// ---------------------------------------------------------------------------
// WMMA fragment loaders (CDNA5 ISA 7.12.2 layouts).
// A (16-bit, 16x32 MxK): lane m=lane&15, g=lane>>4; VGPR j holds K pair at
//   kbase = 2*(j%4) + 16*(j/4) + 8*g.
// B (16-bit, 32x16 KxN, stored as row-major [n][k]): lane n=lane&15, g=lane>>4;
//   VGPR j holds K pair at kbase = 2*(j%4) + 16*(j/4) + 8*g (same as A).
// ---------------------------------------------------------------------------
__device__ __forceinline__ v16h load_frag_a(const half_t* base, int ld, int lane) {
  const int g = (lane >> 4) & 1, m = lane & 15;
  const unsigned int* p = reinterpret_cast<const unsigned int*>(base + m * ld + 8 * g);
  union { v16h h; unsigned int u[8]; } f;
#pragma unroll
  for (int j = 0; j < 4; ++j) f.u[j] = p[j];
#pragma unroll
  for (int j = 0; j < 4; ++j) f.u[4 + j] = p[8 + j];
  return f.h;
}

__device__ __forceinline__ v16h load_frag_b(const half_t* base, int ld, int lane) {
  const int g = (lane >> 4) & 1, n = lane & 15;
  const unsigned int* p = reinterpret_cast<const unsigned int*>(base + n * ld + 8 * g);
  union { v16h h; unsigned int u[8]; } f;
#pragma unroll
  for (int j = 0; j < 4; ++j) f.u[j] = p[j];
#pragma unroll
  for (int j = 0; j < 4; ++j) f.u[4 + j] = p[8 + j];
  return f.h;
}

// ---------------------------------------------------------------------------
// Generic tiled GEMM: Out[m,n] = sum_k A[m,k] * Wt[n,k] + bias[n], epilogues:
//   EPI 0: store f16            EPI 1: exact GELU, store f16
//   EPI 2: f32 out + residual   EPI 3: proj scatter (window-reverse + unshift)
// Block 256 thr = 8 waves (4x2), tile 128x64, each wave 32x32.
// Double-buffered LDS pipeline: one barrier per k-step, global loads for the
// next tile overlap the WMMAs of the current one.
// ---------------------------------------------------------------------------
template <int EPI>
__global__ __launch_bounds__(256) void gemm_wmma_kernel(
    const half_t* __restrict__ A, const half_t* __restrict__ Wt,
    const float* __restrict__ bias, const float* __restrict__ resid,
    void* __restrict__ Out, int M, int K, int Nout) {
  __shared__ __align__(16) half_t Alds[2][128 * 32];
  __shared__ __align__(16) half_t Wlds[2][64 * 32];
  const int tid = threadIdx.x;
  const int lane = tid & 31;
  const int wid = tid >> 5;
  const int wm = wid >> 1, wn = wid & 1;
  const int n0 = blockIdx.x * 64;
  const int m0 = blockIdx.y * 128;
  const int arow = tid >> 2, acq = tid & 3;  // staging: 64 rows x 4 uint4 quarters
  const int nk = K >> 5;

  auto ldA = [&](int kt, int half) {
    return *reinterpret_cast<const uint4*>(
        &A[(size_t)(m0 + arow + half * 64) * K + (kt << 5) + acq * 8]);
  };
  auto ldW = [&](int kt) {
    return *reinterpret_cast<const uint4*>(
        &Wt[(size_t)(n0 + arow) * K + (kt << 5) + acq * 8]);
  };
  auto stA = [&](int buf, int half, uint4 v) {
    *reinterpret_cast<uint4*>(&Alds[buf][(arow + half * 64) * 32 + acq * 8]) = v;
  };
  auto stW = [&](int buf, uint4 v) {
    *reinterpret_cast<uint4*>(&Wlds[buf][arow * 32 + acq * 8]) = v;
  };

  stA(0, 0, ldA(0, 0));
  stA(0, 1, ldA(0, 1));
  stW(0, ldW(0));
  __syncthreads();

  v8f acc[2][2] = {};
  for (int kt = 0; kt < nk; ++kt) {
    const int cur = kt & 1;
    const bool more = (kt + 1) < nk;
    uint4 na0 = make_uint4(0, 0, 0, 0), na1 = na0, nw0 = na0;
    if (more) {
      na0 = ldA(kt + 1, 0);
      na1 = ldA(kt + 1, 1);
      nw0 = ldW(kt + 1);
      if (kt + 2 < nk)  // gfx1250 global_prefetch_b8 two tiles ahead
        __builtin_prefetch(&A[(size_t)(m0 + (tid >> 1)) * K + ((kt + 2) << 5)], 0, 1);
    }
    v16h af[2], bf[2];
    af[0] = load_frag_a(&Alds[cur][(wm * 32 + 0) * 32], 32, lane);
    af[1] = load_frag_a(&Alds[cur][(wm * 32 + 16) * 32], 32, lane);
    bf[0] = load_frag_b(&Wlds[cur][(wn * 32 + 0) * 32], 32, lane);
    bf[1] = load_frag_b(&Wlds[cur][(wn * 32 + 16) * 32], 32, lane);
#pragma unroll
    for (int i = 0; i < 2; ++i)
#pragma unroll
      for (int j = 0; j < 2; ++j) acc[i][j] = WMMA_F16(af[i], bf[j], acc[i][j]);
    if (more) {
      stA(cur ^ 1, 0, na0);
      stA(cur ^ 1, 1, na1);
      stW(cur ^ 1, nw0);
    }
    __syncthreads();
  }

  const int g = (lane >> 4) & 1, cn = lane & 15;
#pragma unroll
  for (int i = 0; i < 2; ++i) {
#pragma unroll
    for (int j = 0; j < 2; ++j) {
      const int col = n0 + wn * 32 + j * 16 + cn;
      const float bv = bias[col];
#pragma unroll
      for (int rj = 0; rj < 8; ++rj) {
        const int row = m0 + wm * 32 + i * 16 + rj + 8 * g;
        float v = acc[i][j][rj] + bv;
        if constexpr (EPI == 1) v = 0.5f * v * (1.0f + erff(v * 0.70710678118654752f));
        if constexpr (EPI == 0 || EPI == 1) {
          reinterpret_cast<half_t*>(Out)[(size_t)row * Nout + col] = (half_t)v;
        } else if constexpr (EPI == 2) {
          const size_t idx = (size_t)row * Nout + col;
          reinterpret_cast<float*>(Out)[idx] = resid[idx] + v;
        } else {  // proj: window reverse + unshift + residual into x2
          const int w = row / 98, t = row % 98;
          const int b = w >> 9, rem = w & 511;
          const int wd = rem >> 6, wh = (rem >> 3) & 7, ww = rem & 7;
          const int td = t / 49, rt = t % 49, th = rt / 7, tw = rt % 7;
          const int d  = (wd * 2 + td + 1) & 15;       // (+SS0) % D
          const int hh = (wh * 7 + th + 3) % 56;       // (+SS1) % H
          const int wc = (ww * 7 + tw + 3) % 56;       // (+SS2) % W
          const size_t idx = ((((size_t)b * 16 + d) * 56 + hh) * 56 + wc) * 256 + col;
          reinterpret_cast<float*>(Out)[idx] = resid[idx] + v;
        }
      }
    }
  }
}

// ---------------------------------------------------------------------------
// Fused window attention: one block per (window, head). 224 thr = 7 waves,
// wave r owns the 16-row tile r of the 112-padded 98-token window.
// bias[head]+mask[widx] are pre-combined into LDS *in the WMMA C-fragment
// layout* (aliasing P) so each lane's 8 adds per tile are one ds_load_b128.
// Softmax runs in registers via shfl_xor over the 16-lane row groups of the
// WMMA C layout. O = P V with K padded to 128.
// ---------------------------------------------------------------------------
__global__ __launch_bounds__(224) void attn_kernel(
    const half_t* __restrict__ qkv, const float* __restrict__ bias_nm,
    const float* __restrict__ mask, half_t* __restrict__ o_win) {
  __shared__ __align__(16) half_t q_lds[112 * 32];
  __shared__ __align__(16) half_t k_lds[112 * 32];
  __shared__ __align__(16) half_t vt_lds[32 * 128];   // V transposed [d][m]
  __shared__ __align__(16) half_t P_lds[112 * 128];   // also aliased as add table
  const int tid = threadIdx.x;
  const int lane = tid & 31;
  const int r = tid >> 5;                 // wave / row tile 0..6
  const int w = blockIdx.x >> 3;          // window 0..1023
  const int head = blockIdx.x & 7;
  const int widx = w & 511;               // mask window index
  const float scale = 0.17677669529663688f;  // 1/sqrt(32)

  for (int li = tid; li < 112 * 4; li += 224) {   // stage q (scaled) and k
    const int row = li >> 2, cq = li & 3;
    union { uint4 v; half_t h[8]; } uq, uk;
    if (row < 98) {
      const size_t base = (size_t)(w * 98 + row) * 768 + head * 32 + cq * 8;
      uq.v = *reinterpret_cast<const uint4*>(&qkv[base]);
      uk.v = *reinterpret_cast<const uint4*>(&qkv[base + 256]);
#pragma unroll
      for (int e = 0; e < 8; ++e) uq.h[e] = (half_t)((float)uq.h[e] * scale);
    } else {
      uq.v = make_uint4(0, 0, 0, 0);
      uk.v = make_uint4(0, 0, 0, 0);
    }
    *reinterpret_cast<uint4*>(&q_lds[row * 32 + cq * 8]) = uq.v;
    *reinterpret_cast<uint4*>(&k_lds[row * 32 + cq * 8]) = uk.v;
  }
  for (int li = tid; li < 128 * 4; li += 224) {   // stage V transposed (uint4
    const int t = li >> 2, dq = li & 3;           // global loads, LDS scatter)
    union { uint4 v; half_t h[8]; } uv;
    if (t < 98)
      uv.v = *reinterpret_cast<const uint4*>(
          &qkv[(size_t)(w * 98 + t) * 768 + 512 + head * 32 + dq * 8]);
    else
      uv.v = make_uint4(0, 0, 0, 0);
#pragma unroll
    for (int e = 0; e < 8; ++e) vt_lds[(dq * 8 + e) * 128 + t] = uv.h[e];
  }
  {  // stage combined bias+mask into C-fragment-swizzled add table
    half_t* add_sw = P_lds;  // alias: consumed before P is produced
    const float* bsrc = bias_nm + (size_t)head * 9604;
    const float* msrc = mask + (size_t)widx * 9604;
    for (int e = tid; e < 112 * 112; e += 224) {
      const int row = e / 112, col = e - row * 112;   // coalesced global reads
      float v = -60000.f;
      if (row < 98 && col < 98) v = bsrc[row * 98 + col] + msrc[row * 98 + col];
      const int r2 = row >> 4, g2 = (row >> 3) & 1, rj = row & 7;
      const int ct = col >> 4, cn2 = col & 15;
      add_sw[(((r2 * 7 + ct) * 32) + g2 * 16 + cn2) * 8 + rj] = (half_t)v;
    }
  }
  __syncthreads();

  v8f acc[7];
  {
    const v16h qf = load_frag_a(&q_lds[(16 * r) * 32], 32, lane);
#pragma unroll
    for (int ct = 0; ct < 7; ++ct) {
      const v16h kf = load_frag_b(&k_lds[(16 * ct) * 32], 32, lane);
      v8f z = {};
      acc[ct] = WMMA_F16(qf, kf, z);
    }
  }
  const int g = (lane >> 4) & 1, cn = lane & 15;
#pragma unroll
  for (int ct = 0; ct < 7; ++ct) {                // + rel-bias + shift mask:
    union { uint4 v; half_t h[8]; } ah;           // one ds_load_b128 per tile
    ah.v = *reinterpret_cast<const uint4*>(&P_lds[(((r * 7 + ct) * 32) + lane) * 8]);
#pragma unroll
    for (int rj = 0; rj < 8; ++rj) acc[ct][rj] += (float)ah.h[rj];
  }
  float inv[8];
#pragma unroll
  for (int rj = 0; rj < 8; ++rj) {                // register softmax per row
    float mx = acc[0][rj];
#pragma unroll
    for (int ct = 1; ct < 7; ++ct) mx = fmaxf(mx, acc[ct][rj]);
#pragma unroll
    for (int off = 8; off >= 1; off >>= 1) mx = fmaxf(mx, __shfl_xor(mx, off, 32));
    float s = 0.f;
#pragma unroll
    for (int ct = 0; ct < 7; ++ct) {
      const float e = __expf(acc[ct][rj] - mx);
      acc[ct][rj] = e;
      s += e;
    }
#pragma unroll
    for (int off = 8; off >= 1; off >>= 1) s += __shfl_xor(s, off, 32);
    inv[rj] = 1.f / s;
  }
  __syncthreads();                                // all add-table reads done
  for (int e = tid; e < 112 * 128 / 8; e += 224)  // zero P (pad rows/cols)
    *reinterpret_cast<uint4*>(&P_lds[e * 8]) = make_uint4(0, 0, 0, 0);
  __syncthreads();
#pragma unroll
  for (int rj = 0; rj < 8; ++rj) {                // write normalized P rows<98
    const int row = 16 * r + rj + 8 * g;
    if (row < 98) {
#pragma unroll
      for (int ct = 0; ct < 7; ++ct)
        P_lds[row * 128 + 16 * ct + cn] = (half_t)(acc[ct][rj] * inv[rj]);
    }
  }
  __syncthreads();

  v8f o[2] = {{}, {}};
#pragma unroll
  for (int ks = 0; ks < 4; ++ks) {                // O = P @ V, K padded 128
    const v16h pf = load_frag_a(&P_lds[(16 * r) * 128 + ks * 32], 128, lane);
#pragma unroll
    for (int nt = 0; nt < 2; ++nt) {
      const v16h vf = load_frag_b(&vt_lds[(16 * nt) * 128 + ks * 32], 128, lane);
      o[nt] = WMMA_F16(pf, vf, o[nt]);
    }
  }
#pragma unroll
  for (int nt = 0; nt < 2; ++nt)
#pragma unroll
    for (int rj = 0; rj < 8; ++rj) {
      const int row = 16 * r + rj + 8 * g;
      if (row < 98)
        o_win[(size_t)(w * 98 + row) * 256 + head * 32 + nt * 16 + cn] = (half_t)o[nt][rj];
    }
}

// ---------------------------------------------------------------------------
// LayerNorm over C=256, optional (roll -shift + window partition) gather.
// One block per token, 256 threads, wave32 shfl_xor reductions.
// ---------------------------------------------------------------------------
__global__ __launch_bounds__(256) void ln_kernel(
    const float* __restrict__ src, const float* __restrict__ wgt,
    const float* __restrict__ bvec, half_t* __restrict__ dst, int windowed) {
  const int m = blockIdx.x;
  const int c = threadIdx.x;
  size_t srcbase;
  if (windowed) {
    const int w = m / 98, t = m % 98;
    const int b = w >> 9, rem = w & 511;
    const int wd = rem >> 6, wh = (rem >> 3) & 7, ww = rem & 7;
    const int td = t / 49, rt = t % 49, th = rt / 7, tw = rt % 7;
    const int d  = (wd * 2 + td + 1) & 15;
    const int hh = (wh * 7 + th + 3) % 56;
    const int wc = (ww * 7 + tw + 3) % 56;
    srcbase = ((((size_t)b * 16 + d) * 56 + hh) * 56 + wc) * 256;
  } else {
    srcbase = (size_t)m * 256;
  }
  const float v = src[srcbase + c];
  const int lane = c & 31, wv = c >> 5;
  __shared__ float part[8];
  float s = v;
#pragma unroll
  for (int off = 16; off >= 1; off >>= 1) s += __shfl_xor(s, off, 32);
  if (lane == 0) part[wv] = s;
  __syncthreads();
  float tot = 0.f;
#pragma unroll
  for (int i = 0; i < 8; ++i) tot += part[i];
  const float mean = tot * (1.f / 256.f);
  const float dv = v - mean;
  float s2 = dv * dv;
#pragma unroll
  for (int off = 16; off >= 1; off >>= 1) s2 += __shfl_xor(s2, off, 32);
  __syncthreads();
  if (lane == 0) part[wv] = s2;
  __syncthreads();
  float tot2 = 0.f;
#pragma unroll
  for (int i = 0; i < 8; ++i) tot2 += part[i];
  const float var = tot2 * (1.f / 256.f);
  dst[(size_t)m * 256 + c] = (half_t)(dv * rsqrtf(var + 1e-5f) * wgt[c] + bvec[c]);
}

__global__ void f32_to_f16_kernel(const float* __restrict__ in, half_t* __restrict__ out, int n) {
  const int i = blockIdx.x * 256 + threadIdx.x;
  if (i < n) out[i] = (half_t)in[i];
}

__global__ void bias_gather_kernel(const float* __restrict__ table, const int* __restrict__ relidx,
                                   float* __restrict__ bias_nm) {
  const int i = blockIdx.x * 256 + threadIdx.x;
  if (i >= 8 * 98 * 98) return;
  const int head = i / 9604, rc = i % 9604;
  bias_nm[i] = table[relidx[rc] * 8 + head];
}

// ---------------------------------------------------------------------------
extern "C" void kernel_launch(void* const* d_in, const int* in_sizes, int n_in,
                              void* d_out, int out_size, void* d_ws, size_t ws_size,
                              hipStream_t stream) {
  (void)in_sizes; (void)n_in; (void)out_size; (void)ws_size;
  const float* x      = (const float*)d_in[0];
  const float* maskm  = (const float*)d_in[1];
  const int*   relidx = (const int*)d_in[2];
  const float* rbt    = (const float*)d_in[3];
  const float* n1w    = (const float*)d_in[4];
  const float* n1b    = (const float*)d_in[5];
  const float* qkv_w  = (const float*)d_in[6];
  const float* qkv_b  = (const float*)d_in[7];
  const float* proj_w = (const float*)d_in[8];
  const float* proj_b = (const float*)d_in[9];
  const float* n2w    = (const float*)d_in[10];
  const float* n2b    = (const float*)d_in[11];
  const float* fc1_w  = (const float*)d_in[12];
  const float* fc1_b  = (const float*)d_in[13];
  const float* fc2_w  = (const float*)d_in[14];
  const float* fc2_b  = (const float*)d_in[15];
  float* out = (float*)d_out;

  const size_t Mrows = 100352;  // B_*N = tokens = 1024*98 = 2*16*56*56
  char* ws = (char*)d_ws;
  size_t off = 0;
  auto alloc = [&](size_t bytes) { size_t r = off; off += (bytes + 255) & ~(size_t)255; return r; };
  half_t* w_qkv16  = (half_t*)(ws + alloc(768 * 256 * 2));
  half_t* w_proj16 = (half_t*)(ws + alloc(256 * 256 * 2));
  half_t* w_fc1_16 = (half_t*)(ws + alloc(1024 * 256 * 2));
  half_t* w_fc2_16 = (half_t*)(ws + alloc(256 * 1024 * 2));
  float*  bias_nm  = (float*)(ws + alloc(8 * 98 * 98 * 4));
  half_t* hbuf     = (half_t*)(ws + alloc(Mrows * 256 * 2));  // h_win -> o_win -> h_ln2
  half_t* qkvb     = (half_t*)(ws + alloc(Mrows * 768 * 2));
  float*  x2       = (float*)(ws + alloc(Mrows * 256 * 4));
  half_t* h2       = (half_t*)(ws + alloc(Mrows * 1024 * 2));

  f32_to_f16_kernel<<<(768 * 256 + 255) / 256, 256, 0, stream>>>(qkv_w, w_qkv16, 768 * 256);
  f32_to_f16_kernel<<<(256 * 256 + 255) / 256, 256, 0, stream>>>(proj_w, w_proj16, 256 * 256);
  f32_to_f16_kernel<<<(1024 * 256 + 255) / 256, 256, 0, stream>>>(fc1_w, w_fc1_16, 1024 * 256);
  f32_to_f16_kernel<<<(256 * 1024 + 255) / 256, 256, 0, stream>>>(fc2_w, w_fc2_16, 256 * 1024);
  bias_gather_kernel<<<(8 * 98 * 98 + 255) / 256, 256, 0, stream>>>(rbt, relidx, bias_nm);

  // LN1 + shift + window partition -> f16 [Mrows, 256]
  ln_kernel<<<Mrows, 256, 0, stream>>>(x, n1w, n1b, hbuf, 1);
  // QKV GEMM: [Mrows,256] x [768,256]^T -> f16 [Mrows,768]
  gemm_wmma_kernel<0><<<dim3(768 / 64, Mrows / 128), 256, 0, stream>>>(
      hbuf, w_qkv16, qkv_b, nullptr, qkvb, (int)Mrows, 256, 768);
  // fused window attention -> o_win f16 [Mrows,256] (reuses hbuf)
  attn_kernel<<<1024 * 8, 224, 0, stream>>>(qkvb, bias_nm, maskm, hbuf);
  // proj + window reverse + unshift + residual -> x2 f32
  gemm_wmma_kernel<3><<<dim3(256 / 64, Mrows / 128), 256, 0, stream>>>(
      hbuf, w_proj16, proj_b, x, x2, (int)Mrows, 256, 256);
  // LN2 -> f16 (reuses hbuf)
  ln_kernel<<<Mrows, 256, 0, stream>>>(x2, n2w, n2b, hbuf, 0);
  // fc1 + GELU -> h2 f16 [Mrows,1024]
  gemm_wmma_kernel<1><<<dim3(1024 / 64, Mrows / 128), 256, 0, stream>>>(
      hbuf, w_fc1_16, fc1_b, nullptr, h2, (int)Mrows, 256, 1024);
  // fc2 + residual -> f32 out
  gemm_wmma_kernel<2><<<dim3(256 / 64, Mrows / 128), 256, 0, stream>>>(
      h2, w_fc2_16, fc2_b, x2, out, (int)Mrows, 1024, 256);
}
